// DeepViT_2233382994661
// MI455X (gfx1250) — matrix operand order repaired and checked by reading, not compiled, not run on hardware
//
#include <hip/hip_runtime.h>

typedef unsigned short u16;
typedef __attribute__((ext_vector_type(16))) __bf16 v16bf;
typedef __attribute__((ext_vector_type(8)))  __bf16 v8bf;
typedef __attribute__((ext_vector_type(8)))  float  v8f;

#if defined(__AMDGCN__) && __has_builtin(__builtin_amdgcn_global_load_async_to_lds_b128)
#define HAVE_ASYNC_LDS 1
typedef int ai4 __attribute__((vector_size(16)));
typedef __attribute__((address_space(1))) ai4* gp_ai4;   // global int4*
typedef __attribute__((address_space(3))) ai4* lp_ai4;   // LDS int4*
#else
#define HAVE_ASYNC_LDS 0
#endif

__device__ __forceinline__ void wait_async_lds() {
#if defined(__AMDGCN__)
#if __has_builtin(__builtin_amdgcn_s_wait_asynccnt)
    __builtin_amdgcn_s_wait_asynccnt(0);
#else
    asm volatile("s_wait_asynccnt 0" ::: "memory");
#endif
#endif
}

__device__ __forceinline__ u16 f2bf(float f) {
    union { float f; unsigned u; } v; v.f = f;
    unsigned r = v.u + 0x7FFFu + ((v.u >> 16) & 1u);
    return (u16)(r >> 16);
}

#define BM 128
#define BN 128
#define BK 64

// C[M,N] = A(bf16,[M,K]) * W(fp32,[K,N]) (+bias) (+resid) (+act), out fp32 or bf16
// resid_mode: 0 none, 1 full [r*N+c], 2 pos-broadcast [(r&63)*N+c]
// act: 0 none, 1 exact GELU, 2 tanh
__global__ __launch_bounds__(256)
void gemm_bf16_wmma(const u16* __restrict__ A, const float* __restrict__ W,
                    const float* __restrict__ bias,
                    const float* __restrict__ resid, int resid_mode,
                    float* __restrict__ Cf, u16* __restrict__ Cb,
                    int M, int N, int K, int act)
{
    __shared__ __attribute__((aligned(32))) u16 sA[BM * BK];   // [m][k]
    __shared__ __attribute__((aligned(32))) u16 sB[BN * BK];   // [n][k] (W transposed)

    const int tid  = threadIdx.x;
    const int lane = tid & 31;
    const int wave = tid >> 5;
    const int m0   = blockIdx.y * BM;
    const int n0   = blockIdx.x * BN;
    const int wm   = (wave >> 1) * 32;
    const int wn   = (wave & 1) * 64;
    const int l15  = lane & 15;
    const int lhi  = lane >> 4;

    v8f acc[2][4];
#pragma unroll
    for (int i = 0; i < 2; ++i)
#pragma unroll
        for (int j = 0; j < 4; ++j) { v8f z = {}; acc[i][j] = z; }

    for (int k0 = 0; k0 < K; k0 += BK) {
        // ---- stage A tile (bf16, row-major copy): 128x64 halves ----
#if HAVE_ASYNC_LDS
#pragma unroll
        for (int t = 0; t < 4; ++t) {
            int chunk = tid + t * 256;             // 1024 chunks of 8 halves
            int row = chunk >> 3;
            int kc  = (chunk & 7) * 8;
            const u16* src = A + (size_t)(m0 + row) * K + k0 + kc;
            u16*       dst = sA + row * BK + kc;
            __builtin_amdgcn_global_load_async_to_lds_b128(
                (gp_ai4)(const_cast<u16*>(src)), (lp_ai4)(dst), 0, 0);
        }
#else
        {
            uint4 r[4];
#pragma unroll
            for (int t = 0; t < 4; ++t) {
                int chunk = tid + t * 256;
                int row = chunk >> 3;
                int kc  = (chunk & 7) * 8;
                r[t] = *(const uint4*)(A + (size_t)(m0 + row) * K + k0 + kc);
            }
#pragma unroll
            for (int t = 0; t < 4; ++t) {
                int chunk = tid + t * 256;
                int row = chunk >> 3;
                int kc  = (chunk & 7) * 8;
                *(uint4*)(sA + row * BK + kc) = r[t];
            }
        }
#endif
        // ---- stage W tile: fp32 [k][n] -> bf16 transposed [n][k] ----
#pragma unroll
        for (int half = 0; half < 2; ++half) {
            float4 v[4];
#pragma unroll
            for (int t = 0; t < 4; ++t) {
                int idx = tid + (half * 4 + t) * 256;   // 2048 float4 total
                int kr  = idx >> 5;
                int nc  = (idx & 31) * 4;
                v[t] = *(const float4*)(W + (size_t)(k0 + kr) * N + n0 + nc);
            }
#pragma unroll
            for (int t = 0; t < 4; ++t) {
                int idx = tid + (half * 4 + t) * 256;
                int kr  = idx >> 5;
                int nc  = (idx & 31) * 4;
                sB[(nc + 0) * BK + kr] = f2bf(v[t].x);
                sB[(nc + 1) * BK + kr] = f2bf(v[t].y);
                sB[(nc + 2) * BK + kr] = f2bf(v[t].z);
                sB[(nc + 3) * BK + kr] = f2bf(v[t].w);
            }
        }
        if (k0 + BK < K) {  // hint next tiles toward L2 (global_prefetch_b8)
            __builtin_prefetch(A + (size_t)(m0 + (tid >> 1)) * K + k0 + BK, 0, 3);
            __builtin_prefetch(W + (size_t)(k0 + BK + (tid >> 2)) * N + n0, 0, 3);
        }
#if HAVE_ASYNC_LDS
        wait_async_lds();      // per-wave: A-tile async copies complete in LDS
#endif
        __syncthreads();

#pragma unroll
        for (int kk = 0; kk < BK; kk += 32) {
            v16bf afr[2], bfr[4];
#pragma unroll
            for (int mi = 0; mi < 2; ++mi) {
                int row = wm + mi * 16 + l15;      // A: M = lane&15
                int kb  = kk + lhi * 16;           // 16 contiguous K per lane
                v8bf lo = *(const v8bf*)(sA + row * BK + kb);       // K=kb..kb+7
                v8bf hi = *(const v8bf*)(sA + row * BK + kb + 8);   // K=kb+8..kb+15
                afr[mi] = __builtin_shufflevector(lo, hi,
                          0,1,2,3,4,5,6,7,8,9,10,11,12,13,14,15);
            }
#pragma unroll
            for (int ni = 0; ni < 4; ++ni) {
                int col = wn + ni * 16 + l15;      // B: N = lane&15
                int kb2 = kk + lhi * 16;           // 16 contiguous K per lane
                bfr[ni] = *(const v16bf*)(sB + col * BK + kb2);
            }
#pragma unroll
            for (int mi = 0; mi < 2; ++mi)
#pragma unroll
                for (int ni = 0; ni < 4; ++ni)
                    acc[mi][ni] = __builtin_amdgcn_wmma_f32_16x16x32_bf16(
                        false, afr[mi], false, bfr[ni],
                        (short)0, acc[mi][ni], false, false);
        }
        __syncthreads();
    }

    // epilogue: D layout — col = lane&15, row = (lane>>4)*8 + vgpr
#pragma unroll
    for (int mi = 0; mi < 2; ++mi) {
#pragma unroll
        for (int ni = 0; ni < 4; ++ni) {
#pragma unroll
            for (int vg = 0; vg < 8; ++vg) {
                int r = m0 + wm + mi * 16 + lhi * 8 + vg;
                int c = n0 + wn + ni * 16 + l15;
                float v = acc[mi][ni][vg];
                if (bias) v += bias[c];
                if (resid_mode == 1)      v += resid[(size_t)r * N + c];
                else if (resid_mode == 2) v += resid[(size_t)(r & 63) * N + c];
                if (act == 1)      v = 0.5f * v * (1.0f + erff(v * 0.7071067811865475f));
                else if (act == 2) v = tanhf(v);
                if (Cb) Cb[(size_t)r * N + c] = f2bf(v);
                else    Cf[(size_t)r * N + c] = v;
            }
        }
    }
}

// Per-row LayerNorm: fp32 in, bf16 out (feeds next GEMM's A operand)
__global__ __launch_bounds__(256)
void layernorm_rows(const float* __restrict__ X, const float* __restrict__ g,
                    const float* __restrict__ b, u16* __restrict__ Y, int D)
{
    __shared__ float red[256];
    int row = blockIdx.x, tid = threadIdx.x;
    const float* x = X + (size_t)row * D;
    float s = 0.f;
    for (int d = tid; d < D; d += 256) s += x[d];
    red[tid] = s; __syncthreads();
    for (int off = 128; off > 0; off >>= 1) {
        if (tid < off) red[tid] += red[tid + off];
        __syncthreads();
    }
    float mu = red[0] / D;
    __syncthreads();
    float q = 0.f;
    for (int d = tid; d < D; d += 256) { float t = x[d] - mu; q += t * t; }
    red[tid] = q; __syncthreads();
    for (int off = 128; off > 0; off >>= 1) {
        if (tid < off) red[tid] += red[tid + off];
        __syncthreads();
    }
    float rs = rsqrtf(red[0] / D + 1e-5f);
    for (int d = tid; d < D; d += 256)
        Y[(size_t)row * D + d] = f2bf((x[d] - mu) * rs * g[d] + b[d]);
}

// img [64,3,256,256] -> patch matrix bf16 [4096, 3072], dim order (p1,p2,c)
__global__ __launch_bounds__(256)
void patchify(const float* __restrict__ img, u16* __restrict__ A0)
{
    int r = blockIdx.x;                 // b*64 + n
    int b = r >> 6, n = r & 63, hh = n >> 3, ww = n & 7;
    for (int d = threadIdx.x; d < 3072; d += 256) {
        int c  = d % 3;
        int pp = d / 3;
        int p1 = pp >> 5, p2 = pp & 31;
        float v = img[(((size_t)(b * 3 + c) * 256) + hh * 32 + p1) * 256 + ww * 32 + p2];
        A0[(size_t)r * 3072 + d] = f2bf(v);
    }
}

// scores + softmax per (b,h): P[(b*8+h), i, j]
__global__ __launch_bounds__(256)
void attn_scores(const float* __restrict__ QKV, float* __restrict__ P)
{
    __shared__ float sq[64 * 64], sk[64 * 64], sd[64 * 64];
    int b = blockIdx.x >> 3, h = blockIdx.x & 7;
    int tid = threadIdx.x;
#pragma unroll
    for (int t = 0; t < 16; ++t) {
        int idx = tid + t * 256; int i = idx >> 6, d = idx & 63;
        size_t base = (size_t)(b * 64 + i) * 1536 + h * 64 + d;
        sq[idx] = QKV[base];
        sk[idx] = QKV[base + 512];
    }
    __syncthreads();
#pragma unroll
    for (int t = 0; t < 16; ++t) {
        int idx = tid + t * 256; int i = idx >> 6, j = idx & 63;
        float s = 0.f;
        for (int d = 0; d < 64; ++d) s += sq[i * 64 + d] * sk[j * 64 + d];
        sd[idx] = s * 0.125f;                    // SCALE = 1/sqrt(64)
    }
    __syncthreads();
    if (tid < 64) {
        int i = tid;
        float m = -1e30f;
        for (int j = 0; j < 64; ++j) m = fmaxf(m, sd[i * 64 + j]);
        float sum = 0.f;
        for (int j = 0; j < 64; ++j) { float e = __expf(sd[i * 64 + j] - m); sd[i * 64 + j] = e; sum += e; }
        float inv = 1.f / sum;
        size_t base = ((size_t)blockIdx.x * 64 + i) * 64;
        for (int j = 0; j < 64; ++j) P[base + j] = sd[i * 64 + j] * inv;
    }
}

// re-attention head mix + LayerNorm over head axis; per (b,i), thread = j
__global__ __launch_bounds__(64)
void attn_remix_ln(const float* __restrict__ P, const float* __restrict__ Wr,
                   const float* __restrict__ g, const float* __restrict__ bb,
                   float* __restrict__ Q)
{
    int bi = blockIdx.x;                 // b*64 + i
    int b = bi >> 6, i = bi & 63;
    int j = threadIdx.x;
    float a[8], mix[8];
#pragma unroll
    for (int h = 0; h < 8; ++h)
        a[h] = P[(((size_t)(b * 8 + h)) * 64 + i) * 64 + j];
    float mu = 0.f;
#pragma unroll
    for (int gg = 0; gg < 8; ++gg) {
        float s = 0.f;
#pragma unroll
        for (int h = 0; h < 8; ++h) s += a[h] * Wr[h * 8 + gg];
        mix[gg] = s; mu += s;
    }
    mu *= 0.125f;
    float var = 0.f;
#pragma unroll
    for (int gg = 0; gg < 8; ++gg) { float t = mix[gg] - mu; var += t * t; }
    var *= 0.125f;
    float rs = rsqrtf(var + 1e-5f);
#pragma unroll
    for (int gg = 0; gg < 8; ++gg)
        Q[(((size_t)(b * 8 + gg)) * 64 + i) * 64 + j] = (mix[gg] - mu) * rs * g[gg] + bb[gg];
}

// out[i,d] = attnLN[i,:] @ V; write bf16 [4096, 512] per (b,h)
__global__ __launch_bounds__(256)
void attn_apply_v(const float* __restrict__ Q, const float* __restrict__ QKV,
                  u16* __restrict__ AO)
{
    __shared__ float sa[64 * 64], sv[64 * 64];
    int b = blockIdx.x >> 3, h = blockIdx.x & 7;
    int tid = threadIdx.x;
#pragma unroll
    for (int t = 0; t < 16; ++t) {
        int idx = tid + t * 256; int i = idx >> 6, d = idx & 63;
        sa[idx] = Q[((size_t)blockIdx.x * 64 + i) * 64 + d];
        sv[idx] = QKV[(size_t)(b * 64 + i) * 1536 + 1024 + h * 64 + d];
    }
    __syncthreads();
#pragma unroll
    for (int t = 0; t < 16; ++t) {
        int idx = tid + t * 256; int i = idx >> 6, d = idx & 63;
        float s = 0.f;
        for (int j = 0; j < 64; ++j) s += sa[i * 64 + j] * sv[j * 64 + d];
        AO[(size_t)(b * 64 + i) * 512 + h * 64 + d] = f2bf(s);
    }
}

// X [b, n, dim] -> out [b, c, 256, 256]
__global__ __launch_bounds__(256)
void unpatchify(const float* __restrict__ X, float* __restrict__ out)
{
    size_t o = (size_t)blockIdx.x * 256 + threadIdx.x;   // over 64*3*256*256
    int x = (int)(o & 255);
    int y = (int)((o >> 8) & 255);
    int bc = (int)(o >> 16);
    int c = bc % 3;
    int b = bc / 3;
    int row = b * 64 + (y >> 5) * 8 + (x >> 5);
    int col = ((y & 31) * 32 + (x & 31)) * 3 + c;
    out[o] = X[(size_t)row * 3072 + col];
}

extern "C" void kernel_launch(void* const* d_in, const int* in_sizes, int n_in,
                              void* d_out, int out_size, void* d_ws, size_t ws_size,
                              hipStream_t stream) {
    (void)in_sizes; (void)n_in; (void)out_size; (void)ws_size;
    const float* img     = (const float*)d_in[0];
    const float* patch_w = (const float*)d_in[1];
    const float* patch_b = (const float*)d_in[2];
    const float* pos_emb = (const float*)d_in[3];
    auto L = [&](int l, int k) { return (const float*)d_in[4 + l * 14 + k]; };
    // per-layer order: 0 ln1_g,1 ln1_b,2 qkv_w,3 reattn_w,4 rln_g,5 rln_b,
    //                  6 out_w,7 out_b,8 ln2_g,9 ln2_b,10 ff1_w,11 ff1_b,12 ff2_w,13 ff2_b

    char* ws = (char*)d_ws;
    size_t off = 0;
    auto alloc = [&](size_t bytes) {
        void* p = ws + off;
        off += (bytes + 255) & ~(size_t)255;
        return p;
    };
    float* X0  = (float*)alloc(4096ull * 3072 * 4);
    float* X1  = (float*)alloc(4096ull * 3072 * 4);
    u16*   Ab  = (u16*)  alloc(4096ull * 3072 * 2);
    u16*   Hb  = (u16*)  alloc(4096ull * 3072 * 2);
    float* QKV = (float*)alloc(4096ull * 1536 * 4);
    float* Pp  = (float*)alloc(512ull * 64 * 64 * 4);
    float* PLN = (float*)alloc(512ull * 64 * 64 * 4);
    u16*   AO  = (u16*)  alloc(4096ull * 512 * 2);

    dim3 blk(256);

    // patch embedding: patches @ patch_w + patch_b + pos_emb (pos via resid_mode=2)
    patchify<<<4096, blk, 0, stream>>>(img, Ab);
    gemm_bf16_wmma<<<dim3(24, 32), blk, 0, stream>>>(
        Ab, patch_w, patch_b, pos_emb, 2, X0, nullptr, 4096, 3072, 3072, 0);

    for (int l = 0; l < 4; ++l) {
        // attention
        layernorm_rows<<<4096, blk, 0, stream>>>(X0, L(l, 0), L(l, 1), Ab, 3072);
        gemm_bf16_wmma<<<dim3(12, 32), blk, 0, stream>>>(
            Ab, L(l, 2), nullptr, nullptr, 0, QKV, nullptr, 4096, 1536, 3072, 0);
        attn_scores<<<512, blk, 0, stream>>>(QKV, Pp);
        attn_remix_ln<<<4096, dim3(64), 0, stream>>>(Pp, L(l, 3), L(l, 4), L(l, 5), PLN);
        attn_apply_v<<<512, blk, 0, stream>>>(PLN, QKV, AO);
        gemm_bf16_wmma<<<dim3(24, 32), blk, 0, stream>>>(
            AO, L(l, 6), L(l, 7), X0, 1, X1, nullptr, 4096, 3072, 512, 0);
        // MLP (bias+GELU fused into ff1, residual+tanh fused into ff2)
        layernorm_rows<<<4096, blk, 0, stream>>>(X1, L(l, 8), L(l, 9), Ab, 3072);
        gemm_bf16_wmma<<<dim3(24, 32), blk, 0, stream>>>(
            Ab, L(l, 10), L(l, 11), nullptr, 0, nullptr, Hb, 4096, 3072, 3072, 1);
        gemm_bf16_wmma<<<dim3(24, 32), blk, 0, stream>>>(
            Hb, L(l, 12), L(l, 13), X1, 1, X0, nullptr, 4096, 3072, 3072, 2);
    }

    unpatchify<<<49152, blk, 0, stream>>>(X0, (float*)d_out);
}
